// SpatialGRU_26766236189197
// MI455X (gfx1250) — matrix operand order repaired and checked by reading, not compiled, not run on hardware
//
#include <hip/hip_runtime.h>
#include <hip/hip_bf16.h>

// Problem constants
#define BB   512
#define CC   8
#define LL   64
#define RR   64
#define UU   32
#define QK   128          // q padded K (104 -> 128)
#define NOUT 256          // 96 (wr) + 128 (wz) + 32 (wij folded)
#define BT   64           // batch tile per workgroup

typedef _Float16 v8h  __attribute__((ext_vector_type(8)));
typedef _Float16 v16h __attribute__((ext_vector_type(16)));
typedef float    v8f  __attribute__((ext_vector_type(8)));

__device__ __forceinline__ v16h cat8(v8h lo, v8h hi) {
  return __builtin_shufflevector(lo, hi, 0,1,2,3,4,5,6,7,8,9,10,11,12,13,14,15);
}
// Contiguous 16 halfs (B-fragment: lane holds K = 16*half .. +15 of its column)
__device__ __forceinline__ v16h ldv16(const _Float16* p) {
  return cat8(*(const v8h*)p, *(const v8h*)(p + 8));
}
// A-fragment: lane holds K = 16*half .. +15 within a 32-wide k-step
__device__ __forceinline__ v16h lda(const _Float16* row, int kbase, int half) {
  return cat8(*(const v8h*)(row + kbase + 16 * half),
              *(const v8h*)(row + kbase + 16 * half + 8));
}

// ---------------------------------------------------------------------------
// Prep: pack weights into f16 combined matrix + bias + U16
//   Wc   [256][128] f16 : rows 0..95 wr_w, 96..223 wz_w, 224..255 wij_w (cols 96..103)
//   bias [256]      f32
//   U16  [32][96]   f16
// ---------------------------------------------------------------------------
__global__ void __launch_bounds__(256) prep_kernel(
    const float* __restrict__ wr_w, const float* __restrict__ wr_b,
    const float* __restrict__ wz_w, const float* __restrict__ wz_b,
    const float* __restrict__ wij_w, const float* __restrict__ wij_b,
    const float* __restrict__ U_w,
    _Float16* __restrict__ Wc, float* __restrict__ biasC,
    _Float16* __restrict__ U16) {
  int idx = blockIdx.x * 256 + threadIdx.x;
  if (idx < NOUT * QK) {
    int n = idx / QK, k = idx % QK;
    float v = 0.f;
    if (k < 104) {
      if (n < 96)        v = wr_w[n * 104 + k];
      else if (n < 224)  v = wz_w[(n - 96) * 104 + k];
      else if (k >= 96)  v = wij_w[(n - 224) * CC + (k - 96)];
    }
    Wc[idx] = (_Float16)v;
  } else if (idx < NOUT * QK + NOUT) {
    int n = idx - NOUT * QK;
    biasC[n] = (n < 96) ? wr_b[n] : (n < 224 ? wz_b[n - 96] : wij_b[n - 224]);
  } else if (idx < NOUT * QK + NOUT + UU * 96) {
    int j = idx - NOUT * QK - NOUT;
    U16[j] = (_Float16)U_w[j];
  }
}

// ---------------------------------------------------------------------------
// One anti-diagonal step. blockIdx.x = cellIdx*8 + batchTile.
// Dynamic LDS layout (124 KB):
//   gates f32 [64][256]   @      0  (64 KB)
//   qh    f16 [64][128]   @  65536  (16 KB)
//   amat  f16 [64][96]    @  81920  (12 KB)
//   hst   f32 [3][64][32] @  94208  (24 KB)  order: top, left, diag
//   hr    f32 [64][32]    @ 118784  ( 8 KB)
// ---------------------------------------------------------------------------
#define SMEM_BYTES 126976

__global__ void __launch_bounds__(256) cell_kernel(
    const float* __restrict__ inputs,
    const _Float16* __restrict__ Wc, const float* __restrict__ biasC,
    const _Float16* __restrict__ U16,
    const float* __restrict__ prev, const float* __restrict__ prev2,
    float* __restrict__ cur, int d, int lmin, float* __restrict__ out) {
  extern __shared__ char smem[];
  float*    gates = (float*)smem;                  // [64][256]
  _Float16* qh    = (_Float16*)(smem + 65536);     // [64][128]
  _Float16* amat  = (_Float16*)(smem + 81920);     // [64][96]
  float*    hst   = (float*)(smem + 94208);        // [3][64][32]
  float*    hr    = (float*)(smem + 118784);       // [64][32]

  const int cellIdx = blockIdx.x >> 3;
  const int mblk    = blockIdx.x & 7;
  const int l  = lmin + cellIdx;
  const int r  = d - l;
  const int b0 = mblk * BT;
  const int tid = threadIdx.x;

  // ---- Phase A1: load neighbor states (guarded boundaries -> 0) ----
  for (int i = tid; i < 3 * BT * UU; i += 256) {
    int which = i / (BT * UU);          // 0=top 1=left 2=diag
    int bu = i - which * (BT * UU);
    int b = bu >> 5, u = bu & 31;
    int gb = b0 + b;
    float v = 0.f;
    if (which == 0)      { if (l > 0)          v = prev [((size_t)(l - 1) * BB + gb) * UU + u]; }
    else if (which == 1) { if (r > 0)          v = prev [((size_t)l       * BB + gb) * UU + u]; }
    else                 { if (l > 0 && r > 0) v = prev2[((size_t)(l - 1) * BB + gb) * UU + u]; }
    hst[i] = v;
  }
  __syncthreads();

  // ---- Phase A2: build q (f16) = [h_top | h_left | h_diag | s_ij | 0pad] ----
  for (int i = tid; i < BT * QK; i += 256) {
    int b = i >> 7, k = i & 127;
    float v = 0.f;
    if (k < 96) {
      v = hst[(k >> 5) * (BT * UU) + b * UU + (k & 31)];
    } else if (k < 104) {
      int c = k - 96;
      v = inputs[(size_t)(b0 + b) * (CC * LL * RR) + (size_t)c * (LL * RR) + l * RR + r];
    }
    qh[i] = (_Float16)v;
  }
  __syncthreads();

  const int wave = tid >> 5, lane = tid & 31;
  const int half = lane >> 4, ln = lane & 15;

  // ---- GEMM1: gates[64,256] = q[64,128] * Wc^T, bias pre-loaded into C ----
  {
    const int n0 = wave * 2;                 // this wave's two N tiles
    float bias0 = biasC[n0 * 16 + ln];
    float bias1 = biasC[(n0 + 1) * 16 + ln];
    v8f acc[4][2];
#pragma unroll
    for (int mt = 0; mt < 4; ++mt)
#pragma unroll
      for (int e = 0; e < 8; ++e) { acc[mt][0][e] = bias0; acc[mt][1][e] = bias1; }

#pragma unroll
    for (int ks = 0; ks < 4; ++ks) {
      const _Float16* w0 = Wc + (size_t)(n0 * 16 + ln) * QK + ks * 32 + 16 * half;
      const _Float16* w1 = Wc + (size_t)((n0 + 1) * 16 + ln) * QK + ks * 32 + 16 * half;
      v16h bf0 = ldv16(w0);
      v16h bf1 = ldv16(w1);
#pragma unroll
      for (int mt = 0; mt < 4; ++mt) {
        v16h af = lda(qh + (mt * 16 + ln) * QK, ks * 32, half);
        acc[mt][0] = __builtin_amdgcn_wmma_f32_16x16x32_f16(
            false, af, false, bf0, (short)0, acc[mt][0], false, false);
        acc[mt][1] = __builtin_amdgcn_wmma_f32_16x16x32_f16(
            false, af, false, bf1, (short)0, acc[mt][1], false, false);
      }
    }
#pragma unroll
    for (int mt = 0; mt < 4; ++mt)
#pragma unroll
      for (int e = 0; e < 8; ++e) {
        int m = mt * 16 + e + 8 * half;
        gates[m * NOUT + n0 * 16 + ln]       = acc[mt][0][e];
        gates[m * NOUT + (n0 + 1) * 16 + ln] = acc[mt][1][e];
      }
  }
  __syncthreads();

  // ---- Phase C: a = sigmoid(rg) * [h_left | h_top | h_diag]  (f16) ----
  for (int i = tid; i < BT * 96; i += 256) {
    int b = i / 96, j = i - b * 96;
    float g  = gates[b * NOUT + j];
    float rg = 1.f / (1.f + __expf(-g));
    int which = (j < 32) ? 1 : (j < 64 ? 0 : 2);   // rg gates [left, top, diag]
    float hv = hst[which * (BT * UU) + b * UU + (j & 31)];
    amat[i] = (_Float16)(rg * hv);
  }
  __syncthreads();

  // ---- GEMM2: hr[64,32] = a[64,96] * U_w^T ----
  {
    const int mt = wave & 3, nt = wave >> 2;      // 8 waves -> 4M x 2N tiles
    v8f acc;
#pragma unroll
    for (int e = 0; e < 8; ++e) acc[e] = 0.f;
#pragma unroll
    for (int ks = 0; ks < 3; ++ks) {
      v16h bf = ldv16(U16 + (size_t)(nt * 16 + ln) * 96 + ks * 32 + 16 * half);
      v16h af = lda(amat + (mt * 16 + ln) * 96, ks * 32, half);
      acc = __builtin_amdgcn_wmma_f32_16x16x32_f16(
          false, af, false, bf, (short)0, acc, false, false);
    }
#pragma unroll
    for (int e = 0; e < 8; ++e)
      hr[(mt * 16 + e + 8 * half) * UU + nt * 16 + ln] = acc[e];
  }
  __syncthreads();

  // ---- Phase E: softmax over {zi,zl,zt,zd}, h_new = tanh(h_l + h_r), blend ----
  const bool last = (l == LL - 1) && (r == RR - 1);
  for (int i = tid; i < BT * UU; i += 256) {
    int b = i >> 5, u = i & 31;
    float zi = gates[b * NOUT +  96 + u];
    float zl = gates[b * NOUT + 128 + u];
    float zt = gates[b * NOUT + 160 + u];
    float zd = gates[b * NOUT + 192 + u];
    float mx = fmaxf(fmaxf(zi, zl), fmaxf(zt, zd));
    float ei = __expf(zi - mx), el = __expf(zl - mx);
    float et = __expf(zt - mx), ed = __expf(zd - mx);
    float inv = 1.f / (ei + el + et + ed);
    float hl_in = gates[b * NOUT + 224 + u];       // s_ij @ wij^T + wij_b (folded)
    float hnew  = tanhf(hl_in + hr[i]);
    float htop  = hst[i];                          // which 0
    float hleft = hst[BT * UU + i];                // which 1
    float hdiag = hst[2 * BT * UU + i];            // which 2
    float h = (el * hleft + et * htop + ed * hdiag + ei * hnew) * inv;
    int gb = b0 + b;
    cur[((size_t)l * BB + gb) * UU + u] = h;
    if (last) out[gb * UU + u] = h;
  }
}

// ---------------------------------------------------------------------------
// Host launch: prep once, then 127 wavefront diagonals with rotating buffers.
// Workspace: Wc(64KB) | bias(1KB) | U16(6KB) | 3 x diag state buffers (4MB ea)
// ---------------------------------------------------------------------------
extern "C" void kernel_launch(void* const* d_in, const int* in_sizes, int n_in,
                              void* d_out, int out_size, void* d_ws, size_t ws_size,
                              hipStream_t stream) {
  const float* inputs = (const float*)d_in[0];
  const float* wr_w   = (const float*)d_in[1];
  const float* wr_b   = (const float*)d_in[2];
  const float* wz_w   = (const float*)d_in[3];
  const float* wz_b   = (const float*)d_in[4];
  const float* wij_w  = (const float*)d_in[5];
  const float* wij_b  = (const float*)d_in[6];
  const float* U_w    = (const float*)d_in[7];

  char* ws = (char*)d_ws;
  _Float16* Wc    = (_Float16*)ws;                        // 65536 B
  float*    biasC = (float*)(ws + 65536);                 //  1024 B
  _Float16* U16   = (_Float16*)(ws + 66560);              //  6144 B
  float* bufs[3];
  const size_t bufElems = (size_t)LL * BB * UU;           // 64*512*32 f32 = 4 MB
  bufs[0] = (float*)(ws + 73728);
  bufs[1] = bufs[0] + bufElems;
  bufs[2] = bufs[1] + bufElems;

  prep_kernel<<<141, 256, 0, stream>>>(wr_w, wr_b, wz_w, wz_b, wij_w, wij_b,
                                       U_w, Wc, biasC, U16);

  for (int d = 0; d < LL + RR - 1; ++d) {
    int lmin = d - (RR - 1); if (lmin < 0) lmin = 0;
    int lmax = d < (LL - 1) ? d : (LL - 1);
    int ncells = lmax - lmin + 1;
    float* curb  = bufs[d % 3];
    float* prevb = bufs[(d + 2) % 3];
    float* prev2 = bufs[(d + 1) % 3];
    dim3 grid(ncells * (BB / BT));
    cell_kernel<<<grid, 256, SMEM_BYTES, stream>>>(
        inputs, Wc, biasC, U16, prevb, prev2, curb, d, lmin, (float*)d_out);
  }
}
